// Decoder_49581102465472
// MI455X (gfx1250) — matrix operand order repaired and checked by reading, not compiled, not run on hardware
//
#include <hip/hip_runtime.h>

// Problem constants
#define Bn 512
#define Tn 1024
#define Fn 64
#define Hn 512
#define G4H 2048   // 4*H
#define KX 576     // F + H (step GEMM K)

typedef __attribute__((ext_vector_type(16))) __bf16 bf16x16;
typedef __attribute__((ext_vector_type(8)))  float  f32x8;
typedef __attribute__((ext_vector_type(4)))  unsigned int u32x4;
typedef __attribute__((ext_vector_type(4)))  int i32x4;
typedef __attribute__((ext_vector_type(8)))  int i32x8;

union Frag32B { bf16x16 v; uint4 q[2]; };
union Acc8    { f32x8 v;   float f[8]; };

#if defined(__has_builtin)
#if __has_builtin(__builtin_amdgcn_tensor_load_to_lds) && \
    __has_builtin(__builtin_amdgcn_s_wait_tensorcnt)
#define HAVE_TDM 1
#endif
#endif
#ifndef HAVE_TDM
#define HAVE_TDM 0
#endif

__device__ __forceinline__ unsigned short f2bf(float x) {
  unsigned int u = __float_as_uint(x);
  u += 0x7fffu + ((u >> 16) & 1u);     // round-to-nearest-even
  return (unsigned short)(u >> 16);
}
__device__ __forceinline__ float sigm(float x)  { return 1.0f / (1.0f + __expf(-x)); }
__device__ __forceinline__ float tanhf_(float x){ return 2.0f * sigm(2.0f * x) - 1.0f; }

// ---- WMMA fragment loaders -------------------------------------------------
// A (16x32 bf16, row-major source): lane<16 -> K 0..15,
// lane>=16 -> K 16..31, M = lane&15
__device__ __forceinline__ bf16x16
load_a_frag(const unsigned short* A, int lda, int row, int kc, int lane) {
  const int kh = (lane >> 4) << 4;                 // 0 or 16
  const unsigned short* p = A + (size_t)row * lda + kc + kh;
  Frag32B f;
  f.q[0] = *reinterpret_cast<const uint4*>(p);
  f.q[1] = *reinterpret_cast<const uint4*>(p + 8);
  return f.v;
}
// B (32x16 bf16) from packed order: blk[((k>>5)*N + n)*32 + (k&31)]
// lanes 0-15 hold K0..15 of col n=lane; lanes 16-31 hold K16..31 of col lane-16
__device__ __forceinline__ bf16x16
load_b_frag(const unsigned short* Bp, int N, int kc, int col, int lane) {
  const unsigned short* p =
      Bp + ((size_t)(kc >> 5) * N + col) * 32 + ((lane >> 4) << 4);
  Frag32B f;
  f.q[0] = *reinterpret_cast<const uint4*>(p);
  f.q[1] = *reinterpret_cast<const uint4*>(p + 8);
  return f.v;
}
// Same layout but chunk-local (LDS staged): base holds one 64-col x 32-k chunk
__device__ __forceinline__ bf16x16
load_b_lds(const unsigned short* base, int col, int lane) {
  const unsigned short* p = base + col * 32 + ((lane >> 4) << 4);
  Frag32B f;
  f.q[0] = *reinterpret_cast<const uint4*>(p);
  f.q[1] = *reinterpret_cast<const uint4*>(p + 8);
  return f.v;
}
__device__ __forceinline__ f32x8 wmma_bf16(bf16x16 a, bf16x16 b, f32x8 c) {
  return __builtin_amdgcn_wmma_f32_16x16x32_bf16(false, a, false, b,
                                                 (short)0, c, false, false);
}

// ---- TDM: 1-D contiguous copy global -> LDS (D# per ISA 08_async_tensor §8)
#if HAVE_TDM
__device__ __forceinline__ void tdm_load_1d(unsigned lds_addr,
                                            const void* gptr,
                                            unsigned bytes) {
  const unsigned long long ga = (unsigned long long)(size_t)gptr;
  const unsigned elems = bytes >> 2;               // 4-byte units
  u32x4 g0;
  g0.x = 1u;                                       // count=1 (valid), user mode
  g0.y = lds_addr;                                 // lds_addr [63:32]
  g0.z = (unsigned)ga;                             // global_addr [95:64]
  g0.w = (unsigned)((ga >> 32) & 0x01ffffffu) | (2u << 30);  // [120:96] | type=2
  union { i32x8 v; unsigned long long q[4]; } g1;
  g1.q[0] = (2ull << 16)                                   // data_size = 4B
          | ((unsigned long long)(elems & 0xffffu) << 48); // tensor_dim0 lo16
  g1.q[1] = (unsigned long long)(elems >> 16)              // tensor_dim0 hi16
          | (1ull << 16)                                   // tensor_dim1 = 1
          | ((unsigned long long)(elems & 0xffffu) << 48); // tile_dim0
  g1.q[2] = ((unsigned long long)elems << 32);             // dim0_stride lo32
  g1.q[3] = 0;                                             // dim0_stride hi, dim1_stride
  const i32x4 z4 = {0, 0, 0, 0};
#if __clang_major__ >= 23
  const i32x8 z8 = {0, 0, 0, 0, 0, 0, 0, 0};
  __builtin_amdgcn_tensor_load_to_lds(g0, g1.v, z4, z4, z8, 0);
#else
  __builtin_amdgcn_tensor_load_to_lds(g0, g1.v, z4, z4, 0);
#endif
}
#endif

// Stage one 64-col x 32-k packed B chunk (4 KB contiguous) into LDS.
__device__ __forceinline__ void stage_b_chunk(unsigned short* dst,
                                              const unsigned short* src,
                                              int wave, int tid) {
#if HAVE_TDM
  (void)tid;
  // each of 4 waves moves its own contiguous 1 KB quarter via the TDM
  tdm_load_1d((unsigned)(size_t)(dst + wave * 512), src + wave * 512, 1024);
#else
  (void)wave;
  uint4* d = reinterpret_cast<uint4*>(dst);
  const uint4* s = reinterpret_cast<const uint4*>(src);
  d[tid] = s[tid];
  d[tid + 128] = s[tid + 128];
#endif
}
__device__ __forceinline__ void wait_b_chunk(int pending) {
#if HAVE_TDM
  if (pending) __builtin_amdgcn_s_wait_tensorcnt(1);
  else         __builtin_amdgcn_s_wait_tensorcnt(0);
#else
  (void)pending;
#endif
}

// ---- Weight repack: Bp[((k>>5)*N + n)*32 + (k&31)] = bf16(W[n][col0+k]) ----
__global__ __launch_bounds__(256)
void pack_b_kernel(const float* __restrict__ W, int ld, int col0,
                   unsigned short* __restrict__ Bp, int N, int K) {
  int i = blockIdx.x * 256 + threadIdx.x;          // over N*K
  if (i >= N * K) return;
  int n = i / K, k = i - n * K;
  Bp[((size_t)(k >> 5) * N + n) * 32 + (k & 31)] =
      f2bf(W[(size_t)n * ld + col0 + k]);
}

__global__ __launch_bounds__(256)
void f32_to_bf16_kernel(const float* __restrict__ s,
                        unsigned short* __restrict__ d, int n) {
  int i = blockIdx.x * 256 + threadIdx.x;
  if (i < n) d[i] = f2bf(s[i]);
}

// ---- Generic bf16 WMMA GEMM: C[M,N] = A[M,K]@Bp + Cinit + bias0 + bias1 ----
// 128 threads (4 waves), 64x64 tile, wave = 32x32 patch (2x2 WMMA).
// B k-chunks staged to LDS by the TDM, double-buffered.
__global__ __launch_bounds__(128)
void gemm_bf16_wmma(const unsigned short* __restrict__ A, int lda,
                    const unsigned short* __restrict__ Bp,
                    const float* __restrict__ Cinit,
                    const float* __restrict__ bias0,
                    const float* __restrict__ bias1,
                    float* __restrict__ C, int N, int K) {
  __shared__ __align__(16) unsigned short ldsB[2][64 * 32];  // 2 x 4 KB
  const int tid  = threadIdx.x;
  const int lane = tid & 31;
  const int wave = tid >> 5;                       // 0..3
  const int blockN = blockIdx.x * 64;
  const int tm  = blockIdx.y * 64 + (wave >> 1) * 32;
  const int tnl = (wave & 1) * 32;                 // block-local col base
  const int cn = lane & 15;
  const int rbase = (lane >> 4) << 3;

  Acc8 acc[2][2];
  for (int mi = 0; mi < 2; ++mi)
    for (int ni = 0; ni < 2; ++ni) {
      const int cc = blockN + tnl + ni * 16 + cn;
      float bv = 0.0f;
      if (bias0) bv += bias0[cc];
      if (bias1) bv += bias1[cc];
      for (int r = 0; r < 8; ++r) {
        float v = bv;
        if (Cinit) v += Cinit[(size_t)(tm + mi * 16 + rbase + r) * N + cc];
        acc[mi][ni].f[r] = v;
      }
    }

  stage_b_chunk(&ldsB[0][0], Bp + (size_t)blockN * 32, wave, tid);
  for (int kc = 0; kc < K; kc += 32) {
    const int buf = (kc >> 5) & 1;
    if (kc + 32 < K) {
      stage_b_chunk(&ldsB[buf ^ 1][0],
                    Bp + ((size_t)((kc + 32) >> 5) * N + blockN) * 32, wave, tid);
      wait_b_chunk(1);
    } else {
      wait_b_chunk(0);
    }
    __syncthreads();                               // chunk visible to all waves
    bf16x16 a0 = load_a_frag(A, lda, tm + cn,      kc, lane);
    bf16x16 a1 = load_a_frag(A, lda, tm + 16 + cn, kc, lane);
    bf16x16 b0 = load_b_lds(&ldsB[buf][0], tnl + cn,      lane);
    bf16x16 b1 = load_b_lds(&ldsB[buf][0], tnl + 16 + cn, lane);
    acc[0][0].v = wmma_bf16(a0, b0, acc[0][0].v);
    acc[0][1].v = wmma_bf16(a0, b1, acc[0][1].v);
    acc[1][0].v = wmma_bf16(a1, b0, acc[1][0].v);
    acc[1][1].v = wmma_bf16(a1, b1, acc[1][1].v);
    __syncthreads();                               // reads done before overwrite
  }
  for (int mi = 0; mi < 2; ++mi)
    for (int ni = 0; ni < 2; ++ni)
      for (int r = 0; r < 8; ++r)
        C[(size_t)(tm + mi * 16 + rbase + r) * N + blockN + tnl + ni * 16 + cn] =
            acc[mi][ni].f[r];
}

// ---- Initial state: h0 = cell0 = tanh(h0pre); x = [y0=0 | h0] --------------
__global__ __launch_bounds__(256)
void init_state_kernel(const float* __restrict__ h0pre,
                       float* __restrict__ cell,
                       unsigned short* __restrict__ xbuf) {
  int i = blockIdx.x * 256 + threadIdx.x;          // over B*H
  if (i >= Bn * Hn) return;
  int b = i >> 9, j = i & (Hn - 1);
  float h = tanhf_(h0pre[i]);
  cell[i] = h;
  xbuf[(size_t)b * KX + Fn + j] = f2bf(h);
  if (j < Fn) xbuf[(size_t)b * KX + j] = 0;        // y0 = 0 (bf16 zero)
}

// ---- Fused LSTM cell update + y = h@W2 + yc + output store ----------------
// 256 threads (8 waves), 32 batch rows per block, grid = 16 blocks
__global__ __launch_bounds__(256)
void lstm_step_kernel(const float* __restrict__ gates,   // [B, 4H]
                      float* __restrict__ cell,          // [B, H]
                      unsigned short* __restrict__ xbuf, // [B, KX]
                      const unsigned short* __restrict__ W2p, // packed h->y
                      const float* __restrict__ yc,      // [B, Fn]
                      float* __restrict__ outp,          // d_out + t*Fn
                      int rowStride) {                   // Tn*Fn
  __shared__ __align__(16) unsigned short hsh[32 * Hn];  // 32 KB bf16 h tile
  const int r0 = blockIdx.x * 32;

  for (int idx = threadIdx.x; idx < 32 * Hn; idx += 256) {
    const int r = idx >> 9, j = idx & (Hn - 1);
    const int b = r0 + r;
    const float* g = gates + (size_t)b * G4H;
    float ig = sigm(g[j]);
    float fg = sigm(g[Hn + j]);
    float gg = tanhf_(g[2 * Hn + j]);
    float og = sigm(g[3 * Hn + j]);
    float cp = cell[(size_t)b * Hn + j];
    float cnew = fg * cp + ig * gg;
    float h = og * tanhf_(cnew);
    cell[(size_t)b * Hn + j] = cnew;
    unsigned short hb = f2bf(h);
    hsh[(r << 9) + j] = hb;
    xbuf[(size_t)b * KX + Fn + j] = hb;              // h part of next x
  }
  __syncthreads();

  // y GEMM: [32 x 64], K = 512, A from LDS, one 16x16 WMMA tile per wave
  const int lane = threadIdx.x & 31;
  const int wave = threadIdx.x >> 5;                 // 0..7
  const int tm = (wave >> 2) * 16;                   // 0 or 16
  const int tn = (wave & 3) * 16;                    // 0..48
  const int cn = lane & 15;
  const int rbase = (lane >> 4) << 3;

  Acc8 acc;
  for (int r = 0; r < 8; ++r)
    acc.f[r] = yc[(size_t)(r0 + tm + rbase + r) * Fn + tn + cn];
  for (int kc = 0; kc < Hn; kc += 32) {
    bf16x16 a = load_a_frag(hsh, Hn, tm + cn, kc, lane);   // ds_load path
    bf16x16 b = load_b_frag(W2p, Fn, kc, tn + cn, lane);
    acc.v = wmma_bf16(a, b, acc.v);
  }
  for (int r = 0; r < 8; ++r) {
    const int row = r0 + tm + rbase + r;
    const int col = tn + cn;
    float y = acc.f[r];
    outp[(size_t)row * rowStride + col] = y;         // out[b, t, :]
    xbuf[(size_t)row * KX + col] = f2bf(y);          // y part of next x
  }
}

// ---------------------------------------------------------------------------
extern "C" void kernel_launch(void* const* d_in, const int* in_sizes, int n_in,
                              void* d_out, int out_size, void* d_ws,
                              size_t ws_size, hipStream_t stream) {
  (void)in_sizes; (void)n_in; (void)out_size; (void)ws_size;
  const float* c     = (const float*)d_in[0];
  // d_in[1] = target (unused: temperature==0 -> no teacher forcing)
  // d_in[2] = temperature (unused)
  const float* V_w   = (const float*)d_in[3];
  const float* V_b   = (const float*)d_in[4];
  const float* W_ih  = (const float*)d_in[5];
  const float* W_hh  = (const float*)d_in[6];
  const float* b_ih  = (const float*)d_in[7];
  const float* b_hh  = (const float*)d_in[8];
  const float* h2o_w = (const float*)d_in[9];
  const float* h2o_b = (const float*)d_in[10];
  float* out = (float*)d_out;

  char* ws = (char*)d_ws;
  auto carve = [&](size_t bytes) {
    char* p = ws; ws += (bytes + 255) & ~(size_t)255; return p;
  };
  unsigned short* Wcp   = (unsigned short*)carve((size_t)KX * G4H * 2);  // [y|h]->gates
  unsigned short* Wccp  = (unsigned short*)carve((size_t)Hn * G4H * 2);  // c->gates
  unsigned short* W2p   = (unsigned short*)carve((size_t)Hn * Fn * 2);   // h->y
  unsigned short* W2cp  = (unsigned short*)carve((size_t)Hn * Fn * 2);   // c->y
  unsigned short* Vtp   = (unsigned short*)carve((size_t)Hn * Hn * 2);   // c->h0
  unsigned short* cbf   = (unsigned short*)carve((size_t)Bn * Hn * 2);
  unsigned short* xbuf  = (unsigned short*)carve((size_t)Bn * KX * 2);
  float* h0pre = (float*)carve((size_t)Bn * Hn * 4);
  float* Gc    = (float*)carve((size_t)Bn * G4H * 4);
  float* ycb   = (float*)carve((size_t)Bn * Fn * 4);
  float* gates = (float*)carve((size_t)Bn * G4H * 4);
  float* cellb = (float*)carve((size_t)Bn * Hn * 4);

  // --- one-time setup: repack weights into WMMA B-fragment order ----------
  auto packGrid = [](int N, int K) { return dim3((N * K + 255) / 256); };
  pack_b_kernel<<<packGrid(G4H, Fn), 256, 0, stream>>>(W_ih, KX, 0,  Wcp,              G4H, Fn);
  pack_b_kernel<<<packGrid(G4H, Hn), 256, 0, stream>>>(W_hh, Hn, 0,  Wcp + (size_t)Fn * G4H, G4H, Hn);
  pack_b_kernel<<<packGrid(G4H, Hn), 256, 0, stream>>>(W_ih, KX, Fn, Wccp,             G4H, Hn);
  pack_b_kernel<<<packGrid(Fn,  Hn), 256, 0, stream>>>(h2o_w, 2 * Hn, 0,  W2p,  Fn, Hn);
  pack_b_kernel<<<packGrid(Fn,  Hn), 256, 0, stream>>>(h2o_w, 2 * Hn, Hn, W2cp, Fn, Hn);
  pack_b_kernel<<<packGrid(Hn,  Hn), 256, 0, stream>>>(V_w, Hn, 0, Vtp, Hn, Hn);
  f32_to_bf16_kernel<<<(Bn * Hn + 255) / 256, 256, 0, stream>>>(c, cbf, Bn * Hn);

  // --- one-time GEMMs: h0pre, Gc, yc --------------------------------------
  gemm_bf16_wmma<<<dim3(Hn / 64, Bn / 64), 128, 0, stream>>>(
      cbf, Hn, Vtp, nullptr, V_b, nullptr, h0pre, Hn, Hn);
  gemm_bf16_wmma<<<dim3(G4H / 64, Bn / 64), 128, 0, stream>>>(
      cbf, Hn, Wccp, nullptr, b_ih, b_hh, Gc, G4H, Hn);
  gemm_bf16_wmma<<<dim3(Fn / 64, Bn / 64), 128, 0, stream>>>(
      cbf, Hn, W2cp, nullptr, h2o_b, nullptr, ycb, Fn, Hn);
  init_state_kernel<<<(Bn * Hn + 255) / 256, 256, 0, stream>>>(h0pre, cellb, xbuf);

  // --- T sequential steps ---------------------------------------------------
  for (int t = 0; t < Tn; ++t) {
    gemm_bf16_wmma<<<dim3(G4H / 64, Bn / 64), 128, 0, stream>>>(
        xbuf, KX, Wcp, Gc, nullptr, nullptr, gates, G4H, KX);
    lstm_step_kernel<<<Bn / 32, 256, 0, stream>>>(
        gates, cellb, xbuf, W2p, ycb, out + (size_t)t * Fn, Tn * Fn);
  }
}
